// AntisymetricRNN_14413910245614
// MI455X (gfx1250) — compile-verified, matrix operand order checked by reading, not run on hardware
//
#include <hip/hip_runtime.h>

// Antisymmetric RNN for MI455X (gfx1250, wave32, WMMA bf16 path).
// 16 workgroups x 16 batch rows (batch rows are independent), each runs the
// full 512-step recurrence. A (1024x1024) is pre-materialized bf16-transposed
// (row = output col, contiguous K) and streamed from L2 each step
// (2MB << 192MB L2). The tanh(h) activation tile (16x1024 bf16, 32KB) is
// shared in LDS across the 16 waves of the workgroup; two s_barriers/step.
// Opaque zero defeats LICM so B-fragments stay as in-loop global_load_b128.
// tanh is a branch-free v_exp_f32/v_rcp_f32 sequence (libm tanhf is branchy
// and was generating ~2.5k VALU/SALU ops per step).

#define H_DIM 1024
#define B_DIM 256
#define D_DIM 256
#define S_LEN 512
#define INV_TAU 0.1f
#define GAMMA_C 0.1f

typedef __attribute__((ext_vector_type(16))) __bf16 v16bf;
typedef __attribute__((ext_vector_type(8)))  __bf16 v8bf;
typedef __attribute__((ext_vector_type(8)))  float  v8f;

static __device__ __forceinline__ v16bf cat8(v8bf lo, v8bf hi) {
    return __builtin_shufflevector(lo, hi, 0,1,2,3,4,5,6,7,8,9,10,11,12,13,14,15);
}

// Non-temporal b128 pair -> one 16-wide bf16 WMMA operand fragment.
static __device__ __forceinline__ v16bf load_frag_nt(const __bf16* p) {
    v8bf lo = __builtin_nontemporal_load((const v8bf*)p);
    v8bf hi = __builtin_nontemporal_load((const v8bf*)(p + 16));
    return cat8(lo, hi);
}

static __device__ __forceinline__ v16bf load_frag_lds(const __bf16* p) {
    v8bf lo = *(const v8bf*)p;
    v8bf hi = *(const v8bf*)(p + 16);
    return cat8(lo, hi);
}

// Branch-free tanh: 1 - 2/(e^{2x}+1), via hardware v_exp_f32 / v_rcp_f32.
// Saturates correctly for |x| large; abs error ~1e-7 (<< bf16 GEMM noise).
static __device__ __forceinline__ float fast_tanh(float x) {
    const float e = __builtin_amdgcn_exp2f(x * 2.885390082f);  // 2*log2(e)
    const float r = __builtin_amdgcn_rcpf(e + 1.0f);
    return __builtin_fmaf(-2.0f, r, 1.0f);
}

// Build At[j][i] = A[i][j] = w_r[i][j] - w_r[j][i], w_r = strict lower tri of v_r.
__global__ __launch_bounds__(256) void prep_At(const float* __restrict__ v_r,
                                               __bf16* __restrict__ At) {
    int idx = blockIdx.x * 256 + threadIdx.x;     // idx = j*H + i
    int j = idx >> 10;
    int i = idx & (H_DIM - 1);
    float a = 0.f;
    if (j < i)      a =  v_r[i * H_DIM + j];
    else if (i < j) a = -v_r[j * H_DIM + i];
    At[idx] = (__bf16)a;
}

__global__ __launch_bounds__(256) void prep_Wt(const float* __restrict__ w_o,
                                               __bf16* __restrict__ Wt) {
    int idx = blockIdx.x * 256 + threadIdx.x;     // w_o is [D][H], already K-contig
    Wt[idx] = (__bf16)w_o[idx];
}

__global__ __launch_bounds__(512, 1) void rnn_main(
    const float* __restrict__ x,       // [S, B, D]
    const float* __restrict__ h_init,  // [B, H]
    const float* __restrict__ b_r,     // [H]
    const float* __restrict__ b_o,     // [D]
    const __bf16* __restrict__ At,     // [H, H]  (A transposed, bf16)
    const __bf16* __restrict__ Wt,     // [D, H]  (w_o, bf16)
    float* __restrict__ out)           // [S, B, D]
{
    __shared__ __bf16 lds_t[16 * H_DIM];   // tanh(h) tile, row-major [m][k], 32KB

    const int lane  = threadIdx.x & 31;
    const int wave  = threadIdx.x >> 5;        // 0..15
    const int b0    = blockIdx.x * 16;         // batch-row tile
    const int cb    = wave * 64;               // this wave's 64 recurrent columns
    const int d0    = wave * 16;               // this wave's 16 output columns
    const int nlo   = lane & 15;               // N (or M) index within tile
    const int hi    = (lane >> 4) & 1;         // lane half: K sub-block / M+8
    const int mrow0 = hi * 8;                  // C/D fragment row base
    const int kb    = hi * 8;                  // A/B fragment K sub-offset

    // ---- persistent h state: 4 tiles of 16 cols, C-fragment layout ----
    v8f   h[4];
    float brv[4];
#pragma unroll
    for (int t = 0; t < 4; ++t) {
        const int j = cb + t * 16 + nlo;
        brv[t] = b_r[j];
#pragma unroll
        for (int r = 0; r < 8; ++r)
            h[t][r] = h_init[(size_t)(b0 + mrow0 + r) * H_DIM + j];
    }
    const float bov = b_o[d0 + nlo];

    // ---- seed LDS with tanh(h_init) ----
#pragma unroll
    for (int t = 0; t < 4; ++t) {
        const int j = cb + t * 16 + nlo;
#pragma unroll
        for (int r = 0; r < 8; ++r)
            lds_t[(mrow0 + r) * H_DIM + j] = (__bf16)fast_tanh(h[t][r]);
    }
    __syncthreads();

    for (int s = 0; s < S_LEN; ++s) {
        // Opaque zero: compiler cannot prove the weight pointers are
        // step-invariant, so the B-fragment loads cannot be hoisted/spilled.
        int zs;
        asm volatile("s_mov_b32 %0, 0" : "=s"(zs));
        const __bf16* At_s = At + zs;
        const __bf16* Wt_s = Wt + zs;

        // ============ recurrent GEMM, two column-tile pairs ============
#pragma unroll
        for (int p = 0; p < 2; ++p) {
            v8f acc0 = {}, acc1 = {};
            const __bf16* bp0 = At_s + (size_t)(cb + (2 * p) * 16 + nlo) * H_DIM + kb;
            const __bf16* bp1 = bp0 + (size_t)16 * H_DIM;
            const __bf16* apt = &lds_t[nlo * H_DIM + kb];
#pragma unroll 2
            for (int k0 = 0; k0 < H_DIM; k0 += 32) {
                v16bf af  = load_frag_lds(apt);
                v16bf bf0 = load_frag_nt(bp0);
                v16bf bf1 = load_frag_nt(bp1);
                acc0 = __builtin_amdgcn_wmma_f32_16x16x32_bf16(
                    false, af, false, bf0, (short)0, acc0, false, false);
                acc1 = __builtin_amdgcn_wmma_f32_16x16x32_bf16(
                    false, af, false, bf1, (short)0, acc1, false, false);
                apt += 32; bp0 += 32; bp1 += 32;
            }
            // h update for tiles 2p, 2p+1 (registers only; LDS untouched)
#pragma unroll
            for (int u = 0; u < 2; ++u) {
                const int t = 2 * p + u;
#pragma unroll
                for (int r = 0; r < 8; ++r) {
                    const float cv = u ? acc1[r] : acc0[r];
                    const float hv = h[t][r];
                    h[t][r] = hv + INV_TAU * fast_tanh(cv + brv[t] - GAMMA_C * hv);
                }
            }
        }

        __syncthreads();   // all waves done reading old tanh tile
#pragma unroll
        for (int t = 0; t < 4; ++t) {
            const int j = cb + t * 16 + nlo;
#pragma unroll
            for (int r = 0; r < 8; ++r)
                lds_t[(mrow0 + r) * H_DIM + j] = (__bf16)fast_tanh(h[t][r]);
        }
        __syncthreads();   // new tanh tile visible to all waves

        // ============ output projection: o = tanh(h_new) @ w_o^T ============
        v8f o = {};
        {
            const __bf16* wp  = Wt_s + (size_t)(d0 + nlo) * H_DIM + kb;
            const __bf16* apt = &lds_t[nlo * H_DIM + kb];
#pragma unroll 2
            for (int k0 = 0; k0 < H_DIM; k0 += 32) {
                v16bf af = load_frag_lds(apt);
                v16bf wf = load_frag_nt(wp);
                o = __builtin_amdgcn_wmma_f32_16x16x32_bf16(
                    false, af, false, wf, (short)0, o, false, false);
                apt += 32; wp += 32;
            }
        }

        // ============ epilogue: err = o + b_o - x_t ============
        const float* xp = &x[((size_t)s * B_DIM + b0) * D_DIM];
        float*       op = &out[((size_t)s * B_DIM + b0) * D_DIM];
        const int d = d0 + nlo;
#pragma unroll
        for (int r = 0; r < 8; ++r) {
            const int m = mrow0 + r;
            const float xv = __builtin_nontemporal_load(&xp[m * D_DIM + d]);
            __builtin_nontemporal_store(o[r] + bov - xv, &op[m * D_DIM + d]);
        }
    }
}

extern "C" void kernel_launch(void* const* d_in, const int* in_sizes, int n_in,
                              void* d_out, int out_size, void* d_ws, size_t ws_size,
                              hipStream_t stream) {
    const float* x      = (const float*)d_in[0];
    const float* h_init = (const float*)d_in[1];
    const float* v_r    = (const float*)d_in[2];
    const float* b_r    = (const float*)d_in[3];
    const float* w_o    = (const float*)d_in[4];
    const float* b_o    = (const float*)d_in[5];
    float* out = (float*)d_out;

    __bf16* At = (__bf16*)d_ws;                                        // 2 MB
    __bf16* Wt = (__bf16*)((char*)d_ws + (size_t)H_DIM * H_DIM * 2);   // 0.5 MB

    prep_At<<<(H_DIM * H_DIM) / 256, 256, 0, stream>>>(v_r, At);
    prep_Wt<<<(D_DIM * H_DIM) / 256, 256, 0, stream>>>(w_o, Wt);
    rnn_main<<<B_DIM / 16, 512, 0, stream>>>(x, h_init, b_r, b_o, At, Wt, out);
}
